// TreeConvLSTM3d_5059471474867
// MI455X (gfx1250) — matrix operand order repaired and checked by reading, not compiled, not run on hardware
//
#include <hip/hip_runtime.h>
#include <hip/hip_bf16.h>
#include <math.h>

typedef __attribute__((ext_vector_type(8)))  _Float16 v8h;
typedef __attribute__((ext_vector_type(16))) _Float16 v16h;
typedef __attribute__((ext_vector_type(8)))  float    v8f;

#define NN 31      // tree nodes
#define CH 10      // lstm hidden channels
#define SV 512     // 8^3 voxels
#define CP 32      // padded channel storage (channels-last f16)

__device__ __forceinline__ float sigf(float x) { return 1.0f / (1.0f + __expf(-x)); }

// ---------------------------------------------------------------------------
// Implicit-GEMM 3x3x3 conv (pad=1), tap-major K loop, v_wmma_f32_16x16x32_f16.
// Activations: f16 channels-last [nb, D^3, 32] (pad channels zero-filled).
// Weights pre-packed f16: wp[co(32)][tap(27)][Kp]  (Kp = nblk*32, zero-padded).
// K block 0 = source0 channels (optionally 2x nearest upsampled), block 1 = source1.
// A-fragment: per lane two contiguous v8h loads (K = hi*16+0..7 , hi*16+8..15).
// B-fragment: one contiguous v16h load (K = hi*16 + e).
// ---------------------------------------------------------------------------
__global__ __launch_bounds__(128) void conv3x3_wmma_kernel(
    const _Float16* __restrict__ in0, int up0,
    const _Float16* __restrict__ in1,
    const _Float16* __restrict__ wp,
    const float* __restrict__ bias,
    _Float16* __restrict__ out,
    int COUT, int D, int nb, int act, int nblk, int zero_hi)
{
    const int lane = threadIdx.x & 31;
    const int wave = threadIdx.x >> 5;
    const int D3   = D * D * D;
    const int Mtiles = (nb * D3) >> 4;
    const int mtile  = blockIdx.x * 4 + wave;
    if (mtile >= Mtiles) return;                     // uniform per-wave exit

    const int hi = lane >> 4;
    const int co = (blockIdx.y << 4) + (lane & 15);  // B/C/D column (< 32)
    const int Kp = nblk << 5;

    // A-fragment row coordinates (one row per lane; D^3 multiple of 16 so a
    // tile never crosses a sample boundary, but decode per-lane anyway).
    const int mA = (mtile << 4) + (lane & 15);
    const int nA = mA / D3;
    const int rA = mA - nA * D3;
    const int zA = rA / (D * D);
    const int tA = rA - zA * D * D;
    const int yA = tA / D;
    const int xA = tA - yA * D;

    const int  Dh = D >> 1;
    const long plane0 = up0 ? (long)Dh * Dh * Dh : (long)D3;
    const _Float16* src0 = in0 + (long)nA * plane0 * CP;
    const _Float16* src1 = (in1 != nullptr) ? (in1 + (long)nA * D3 * CP) : nullptr;
    const _Float16* wrow = wp + (long)co * 27 * Kp;

    v8f acc;
    {
        float bv = (bias != nullptr && co < COUT) ? bias[co] : 0.0f;
        #pragma unroll
        for (int r = 0; r < 8; ++r) acc[r] = bv;
    }

    const v8h zer = {};
    for (int tap = 0; tap < 27; ++tap) {
        const int kd = tap / 9;
        const int r2 = tap - kd * 9;
        const int kh = r2 / 3;
        const int kw = r2 - kh * 3;
        const int zz = zA + kd - 1, yy = yA + kh - 1, xx = xA + kw - 1;
        const bool inb = (unsigned)zz < (unsigned)D && (unsigned)yy < (unsigned)D &&
                         (unsigned)xx < (unsigned)D;

        // ---- source 0 (optionally upsampled) ----
        long vox0 = up0 ? (((long)(zz >> 1) * Dh + (yy >> 1)) * Dh + (xx >> 1))
                        : (((long)zz * D + yy) * D + xx);
        const _Float16* p0 = src0 + (inb ? vox0 : 0) * CP;
        v8h a_lo = *(const v8h*)(p0 + hi * 16);
        v8h a_hi = *(const v8h*)(p0 + hi * 16 + 8);
        if (!inb) { a_lo = zer; a_hi = zer; }
        v16h afrag = __builtin_shufflevector(a_lo, a_hi,
                        0,1,2,3,4,5,6,7,8,9,10,11,12,13,14,15);
        v16h bfrag = *(const v16h*)(wrow + tap * Kp + hi * 16);
        acc = __builtin_amdgcn_wmma_f32_16x16x32_f16(
                  false, afrag, false, bfrag, (short)0, acc, false, false);

        // ---- source 1 (concat tail, never upsampled) ----
        if (nblk == 2) {
            long vox1 = ((long)zz * D + yy) * D + xx;
            const _Float16* p1 = src1 + (inb ? vox1 : 0) * CP;
            v8h c_lo = *(const v8h*)(p1 + hi * 16);
            v8h c_hi = *(const v8h*)(p1 + hi * 16 + 8);
            if (!inb) { c_lo = zer; c_hi = zer; }
            v16h afrag2 = __builtin_shufflevector(c_lo, c_hi,
                            0,1,2,3,4,5,6,7,8,9,10,11,12,13,14,15);
            v16h bfrag2 = *(const v16h*)(wrow + tap * Kp + 32 + hi * 16);
            acc = __builtin_amdgcn_wmma_f32_16x16x32_f16(
                      false, afrag2, false, bfrag2, (short)0, acc, false, false);
        }
    }

    // store: lane column co; VGPR r -> row r + hi*8; pad columns written as 0
    #pragma unroll
    for (int r = 0; r < 8; ++r) {
        int m = (mtile << 4) + r + (hi << 3);
        int n = m / D3;
        int rem = m - n * D3;
        float v = acc[r];
        if (act == 1) v = fmaxf(v, 0.0f);
        else if (act == 2) v = sigf(v);
        long oidx = ((long)n * D3 + rem) * CP + co;
        out[oidx] = (_Float16)v;
        if (zero_hi) out[oidx + 16] = (_Float16)0.0f;
    }
}

// ---------------------------------------------------------------------------
// Weight repack: wp[co(32)][tap(27)][Kp] f16, zero padded.
//   k < 32  -> source0 channel k   (valid if k < C0)
//   k >= 32 -> source1 channel k-32 (valid if k-32 < C1)
// Original w: [COUT][C0+C1][27] f32 (concat order source0 then source1).
// ---------------------------------------------------------------------------
__global__ void pack_weights_kernel(const float* __restrict__ w, _Float16* __restrict__ wpk,
                                    int COUT, int C0, int C1, int Kp)
{
    int i = blockIdx.x * blockDim.x + threadIdx.x;
    int total = 32 * 27 * Kp;
    if (i >= total) return;
    int k   = i % Kp;
    int tap = (i / Kp) % 27;
    int co  = i / (Kp * 27);
    float v = 0.0f;
    if (co < COUT) {
        int ci = -1;
        if (k < 32) { if (k < C0) ci = k; }
        else        { if (k - 32 < C1) ci = C0 + (k - 32); }
        if (ci >= 0) v = w[((long)co * (C0 + C1) + ci) * 27 + tap];
    }
    wpk[i] = (_Float16)v;
}

// ---------------------------------------------------------------------------
// e1a: 1 -> 16ch direct conv (Cin=1 is a poor WMMA fit; 0.9 GF, memory bound).
// data f32 [31,1,32^3] -> out f16 channels-last [31,32^3,32] with ReLU.
// ---------------------------------------------------------------------------
__global__ void conv_e1a_kernel(const float* __restrict__ data, const float* __restrict__ w,
                                const float* __restrict__ b, _Float16* __restrict__ out)
{
    long m = (long)blockIdx.x * blockDim.x + threadIdx.x;
    if (m >= (long)NN * 32768) return;
    int n = (int)(m >> 15);
    int rem = (int)(m & 32767);
    int z = rem >> 10, y = (rem >> 5) & 31, x = rem & 31;
    const float* p = data + (long)n * 32768;
    float nbv[27];
    #pragma unroll
    for (int kd = 0; kd < 3; ++kd)
        #pragma unroll
        for (int kh = 0; kh < 3; ++kh)
            #pragma unroll
            for (int kw = 0; kw < 3; ++kw) {
                int zz = z + kd - 1, yy = y + kh - 1, xx = x + kw - 1;
                bool inb = (unsigned)zz < 32u && (unsigned)yy < 32u && (unsigned)xx < 32u;
                nbv[(kd * 3 + kh) * 3 + kw] = inb ? p[(zz << 10) + (yy << 5) + xx] : 0.0f;
            }
    _Float16* o = out + m * CP;
    #pragma unroll
    for (int co = 0; co < 16; ++co) {
        float s = b[co];
        #pragma unroll
        for (int t = 0; t < 27; ++t) s += nbv[t] * w[co * 27 + t];
        o[co] = (_Float16)fmaxf(s, 0.0f);
    }
    #pragma unroll
    for (int co = 16; co < 32; ++co) o[co] = (_Float16)0.0f;
}

// ---------------------------------------------------------------------------
// 2x2x2 max pool, channels-last. in [nb, D^3, 32] -> out [nb, (D/2)^3, 32]
// ---------------------------------------------------------------------------
__global__ void maxpool2_kernel(const _Float16* __restrict__ in, _Float16* __restrict__ out,
                                int nb, int D)
{
    const int Do = D >> 1;
    const int Do3 = Do * Do * Do;
    long i = (long)blockIdx.x * blockDim.x + threadIdx.x;
    if (i >= (long)nb * Do3 * CP) return;
    int ch = (int)(i & 31);
    long t = i >> 5;
    int vox = (int)(t % Do3);
    int n = (int)(t / Do3);
    int z = vox / (Do * Do);
    int r = vox - z * Do * Do;
    int y = r / Do;
    int x = r - y * Do;
    const _Float16* p = in + (long)n * D * D * D * CP;
    float m = -3.402823466e38f;
    #pragma unroll
    for (int dz = 0; dz < 2; ++dz)
        #pragma unroll
        for (int dy = 0; dy < 2; ++dy)
            #pragma unroll
            for (int dx = 0; dx < 2; ++dx)
                m = fmaxf(m, (float)p[(((long)(2*z+dz) * D + (2*y+dy)) * D + (2*x+dx)) * CP + ch]);
    out[i] = (_Float16)m;
}

// ---------------------------------------------------------------------------
// Leaves (nodes 15..30): iou = Wx + b_iou; c = sig(i)*tanh(u)+c0; h = sig(o)*tanh(c)
// Wx/h/c channels-last f16; b_iou f32 [30,512]; c0 f32 NCDHW [31,10,512].
// ---------------------------------------------------------------------------
__global__ void leaf_kernel(const _Float16* __restrict__ Wx, const float* __restrict__ b_iou,
                            const float* __restrict__ c0,
                            _Float16* __restrict__ h_all, _Float16* __restrict__ c_all)
{
    int i = blockIdx.x * blockDim.x + threadIdx.x;
    if (i >= 16 * SV * CP) return;
    int ch = i & 31;
    int v  = (i >> 5) & (SV - 1);
    int p  = i >> 14;
    int node = 15 + p;
    long vb = ((long)node * SV + v) * CP;
    _Float16 hv = (_Float16)0.0f, cv = (_Float16)0.0f;
    if (ch < CH) {
        float iv = (float)Wx[vb + ch]          + b_iou[ch * SV + v];
        float ov = (float)Wx[vb + CH + ch]     + b_iou[(CH + ch) * SV + v];
        float uv = (float)Wx[vb + 2 * CH + ch] + b_iou[(2 * CH + ch) * SV + v];
        float cin = c0[((long)node * CH + ch) * SV + v];
        float c = sigf(iv) * tanhf(uv) + cin;
        float h = sigf(ov) * tanhf(c);
        hv = (_Float16)h; cv = (_Float16)c;
    }
    h_all[vb + ch] = hv;
    c_all[vb + ch] = cv;
}

// ---------------------------------------------------------------------------
// hsum[p] = h[childFirst+2p] + h[childFirst+2p+1]   (channels-last)
// ---------------------------------------------------------------------------
__global__ void hsum_kernel(const _Float16* __restrict__ h_all, _Float16* __restrict__ hsum,
                            int childFirst, int P)
{
    int i = blockIdx.x * blockDim.x + threadIdx.x;
    if (i >= P * SV * CP) return;
    int p = i >> 14;
    int rem = i & ((SV * CP) - 1);
    long c0i = (long)(childFirst + 2 * p) * SV * CP + rem;
    long c1i = (long)(childFirst + 2 * p + 1) * SV * CP + rem;
    hsum[i] = (_Float16)((float)h_all[c0i] + (float)h_all[c1i]);
}

// ---------------------------------------------------------------------------
// Node update: c = sig(i)*tanh(u) + f0*c_ch0 + f1*c_ch1; h = sig(o)*tanh(c)
// iou [P,512,32] (30ch), f [2P,512,32] (sigmoided), all channels-last f16.
// ---------------------------------------------------------------------------
__global__ void node_update_kernel(const _Float16* __restrict__ iou, const _Float16* __restrict__ f,
                                   const float* __restrict__ b_iou,
                                   _Float16* __restrict__ h_all, _Float16* __restrict__ c_all,
                                   int first, int P)
{
    int i = blockIdx.x * blockDim.x + threadIdx.x;
    if (i >= P * SV * CP) return;
    int ch = i & 31;
    int v  = (i >> 5) & (SV - 1);
    int p  = i >> 14;
    int node = first + p;
    int child0 = 2 * node + 1;
    long vb = ((long)node * SV + v) * CP;
    _Float16 hv = (_Float16)0.0f, cv = (_Float16)0.0f;
    if (ch < CH) {
        long ib = ((long)p * SV + v) * CP;
        float iv = (float)iou[ib + ch]          + b_iou[ch * SV + v];
        float ov = (float)iou[ib + CH + ch]     + b_iou[(CH + ch) * SV + v];
        float uv = (float)iou[ib + 2 * CH + ch] + b_iou[(2 * CH + ch) * SV + v];
        float f0 = (float)f[((long)(2 * p) * SV + v) * CP + ch];
        float f1 = (float)f[((long)(2 * p + 1) * SV + v) * CP + ch];
        float cc0 = (float)c_all[((long)child0 * SV + v) * CP + ch];
        float cc1 = (float)c_all[((long)(child0 + 1) * SV + v) * CP + ch];
        float c = sigf(iv) * tanhf(uv) + f0 * cc0 + f1 * cc1;
        float h = sigf(ov) * tanhf(c);
        hv = (_Float16)h; cv = (_Float16)c;
    }
    h_all[vb + ch] = hv;
    c_all[vb + ch] = cv;
}

// ---------------------------------------------------------------------------
// Final 1x1x1 conv 16->1, pad 0. in f16 channels-last -> out f32 [31,1,32^3]
// ---------------------------------------------------------------------------
__global__ void conv1x1_kernel(const _Float16* __restrict__ in, const float* __restrict__ w,
                               const float* __restrict__ b, float* __restrict__ out)
{
    long i = (long)blockIdx.x * blockDim.x + threadIdx.x;
    if (i >= (long)NN * 32768) return;
    const _Float16* p = in + i * CP;
    float s = b[0];
    #pragma unroll
    for (int c = 0; c < 16; ++c) s += (float)p[c] * w[c];
    out[i] = s;
}

// ---------------------------------------------------------------------------
// Host-side orchestration
// ---------------------------------------------------------------------------
static inline void launch_conv(const _Float16* in0, int up0, const _Float16* in1,
                               const _Float16* wp, const float* b, _Float16* out,
                               int COUT, int D, int nb, int act, int nblk,
                               hipStream_t stream)
{
    int Mtiles = nb * D * D * D / 16;
    int zero_hi = (COUT <= 16) ? 1 : 0;
    dim3 grid((Mtiles + 3) / 4, zero_hi ? 1 : 2);
    conv3x3_wmma_kernel<<<grid, dim3(128), 0, stream>>>(
        in0, up0, in1, wp, b, out, COUT, D, nb, act, nblk, zero_hi);
}

static inline void launch_pack(const float* w, _Float16* wp, int COUT, int C0, int C1,
                               int Kp, hipStream_t stream)
{
    int total = 32 * 27 * Kp;
    pack_weights_kernel<<<(total + 255) / 256, 256, 0, stream>>>(w, wp, COUT, C0, C1, Kp);
}

extern "C" void kernel_launch(void* const* d_in, const int* in_sizes, int n_in,
                              void* d_out, int out_size, void* d_ws, size_t ws_size,
                              hipStream_t stream)
{
    (void)in_sizes; (void)n_in; (void)out_size; (void)ws_size;
    const float* data  = (const float*)d_in[0];
    const float* c0    = (const float*)d_in[2];
    const float* e1a_w = (const float*)d_in[3];  const float* e1a_b = (const float*)d_in[4];
    const float* e1b_w = (const float*)d_in[5];  const float* e1b_b = (const float*)d_in[6];
    const float* e2a_w = (const float*)d_in[7];  const float* e2a_b = (const float*)d_in[8];
    const float* e2b_w = (const float*)d_in[9];  const float* e2b_b = (const float*)d_in[10];
    const float* d2a_w = (const float*)d_in[11]; const float* d2a_b = (const float*)d_in[12];
    const float* d2b_w = (const float*)d_in[13]; const float* d2b_b = (const float*)d_in[14];
    const float* d1a_w = (const float*)d_in[15]; const float* d1a_b = (const float*)d_in[16];
    const float* d1b_w = (const float*)d_in[17]; const float* d1b_b = (const float*)d_in[18];
    const float* d1c_w = (const float*)d_in[19]; const float* d1c_b = (const float*)d_in[20];
    const float* w_iou = (const float*)d_in[21];
    const float* u_iou = (const float*)d_in[22];
    const float* u_f   = (const float*)d_in[23];
    const float* b_iou = (const float*)d_in[24];
    float* out = (float*)d_out;
    _Float16* ws = (_Float16*)d_ws;

    // ---- workspace layout (f16 elements; all offsets 64B-aligned) ----
    const long SZ32 = 31L * 32768 * CP;   // 32^3 tensors
    const long SZ16 = 31L * 4096  * CP;   // 16^3 tensors
    const long SZ8  = 31L * SV    * CP;   // 8^3 tensors
    const long o_t1   = 0;                // e1a out; reused as d1a out (g1)
    const long o_x0   = o_t1  + SZ32;     // skip 1
    const long o_g2   = o_x0  + SZ32;     // d1b out
    const long o_xp1  = o_g2  + SZ32;
    const long o_t2   = o_xp1 + SZ16;     // e2a out; reused as d2a out
    const long o_x1   = o_t2  + SZ16;     // skip 2
    const long o_db   = o_x1  + SZ16;     // d2b out
    const long o_xp2  = o_db  + SZ16;
    const long o_Wx   = o_xp2 + SZ8;
    const long o_h    = o_Wx  + SZ8;
    const long o_c    = o_h   + SZ8;
    const long o_hsum = o_c   + SZ8;                  // <= 8*512*32
    const long o_f    = o_hsum + 8L * SV * CP;        // <= 16*512*32
    const long o_iou  = o_f   + 16L * SV * CP;        // <= 8*512*32
    long       o_wp   = o_iou + 8L * SV * CP;
    const long WPSZ   = 32L * 27 * 64;                // max pack size (Kp=64)
    const long p_e1b = o_wp; o_wp += WPSZ;
    const long p_e2a = o_wp; o_wp += WPSZ;
    const long p_e2b = o_wp; o_wp += WPSZ;
    const long p_wiou = o_wp; o_wp += WPSZ;
    const long p_uiou = o_wp; o_wp += WPSZ;
    const long p_uf   = o_wp; o_wp += WPSZ;
    const long p_d2a = o_wp; o_wp += WPSZ;
    const long p_d2b = o_wp; o_wp += WPSZ;
    const long p_d1a = o_wp; o_wp += WPSZ;
    const long p_d1b = o_wp; o_wp += WPSZ;

    // ---- weight repacks (f32 -> padded f16 [32][27][Kp]) ----
    launch_pack(e1b_w, ws + p_e1b, 16, 16, 0, 32, stream);
    launch_pack(e2a_w, ws + p_e2a, 32, 16, 0, 32, stream);
    launch_pack(e2b_w, ws + p_e2b, 20, 32, 0, 32, stream);
    launch_pack(w_iou, ws + p_wiou, 30, 20, 0, 32, stream);
    launch_pack(u_iou, ws + p_uiou, 30, 10, 0, 32, stream);
    launch_pack(u_f,   ws + p_uf,   10, 10, 0, 32, stream);
    launch_pack(d2a_w, ws + p_d2a, 20, 10, 20, 64, stream);   // concat: up2(h) | x1
    launch_pack(d2b_w, ws + p_d2b, 32, 20, 0, 32, stream);
    launch_pack(d1a_w, ws + p_d1a, 16, 32, 16, 64, stream);   // concat: up2(db) | x0
    launch_pack(d1b_w, ws + p_d1b, 16, 16, 0, 32, stream);

    // ---------------- Encoder ----------------
    {
        long tot = 31L * 32768;
        conv_e1a_kernel<<<(tot + 255) / 256, 256, 0, stream>>>(data, e1a_w, e1a_b, ws + o_t1);
    }
    launch_conv(ws + o_t1, 0, nullptr, ws + p_e1b, e1b_b, ws + o_x0, 16, 32, NN, 1, 1, stream);
    {
        long tot = 31L * 4096 * CP;
        maxpool2_kernel<<<(tot + 255) / 256, 256, 0, stream>>>(ws + o_x0, ws + o_xp1, NN, 32);
    }
    launch_conv(ws + o_xp1, 0, nullptr, ws + p_e2a, e2a_b, ws + o_t2, 32, 16, NN, 1, 1, stream);
    launch_conv(ws + o_t2, 0, nullptr, ws + p_e2b, e2b_b, ws + o_x1, 20, 16, NN, 1, 1, stream);
    {
        long tot = 31L * SV * CP;
        maxpool2_kernel<<<(tot + 255) / 256, 256, 0, stream>>>(ws + o_x1, ws + o_xp2, NN, 16);
    }
    launch_conv(ws + o_xp2, 0, nullptr, ws + p_wiou, nullptr, ws + o_Wx, 30, 8, NN, 0, 1, stream);

    // ---------------- Tree LSTM (bottom-up) ----------------
    {
        int tot = 16 * SV * CP;
        leaf_kernel<<<(tot + 255) / 256, 256, 0, stream>>>(ws + o_Wx, b_iou, c0,
                                                           ws + o_h, ws + o_c);
    }
    for (int lvl = 3; lvl >= 0; --lvl) {
        int first = (1 << lvl) - 1;
        int P = 1 << lvl;
        int childFirst = 2 * first + 1;
        // f = sigmoid(conv(h_children, u_f))  [2P,512,32]
        launch_conv(ws + o_h + (long)childFirst * SV * CP, 0, nullptr,
                    ws + p_uf, nullptr, ws + o_f, 10, 8, 2 * P, 2, 1, stream);
        {
            int tot = P * SV * CP;
            hsum_kernel<<<(tot + 255) / 256, 256, 0, stream>>>(ws + o_h, ws + o_hsum,
                                                               childFirst, P);
        }
        // iou = conv(hsum, u_iou)  [P,512,32]
        launch_conv(ws + o_hsum, 0, nullptr, ws + p_uiou, nullptr, ws + o_iou,
                    30, 8, P, 0, 1, stream);
        {
            int tot = P * SV * CP;
            node_update_kernel<<<(tot + 255) / 256, 256, 0, stream>>>(
                ws + o_iou, ws + o_f, b_iou, ws + o_h, ws + o_c, first, P);
        }
    }

    // ---------------- Decoder (up2 + concat fused into conv A-fetch) ----------------
    launch_conv(ws + o_h, 1, ws + o_x1, ws + p_d2a, d2a_b, ws + o_t2, 20, 16, NN, 1, 2, stream);
    launch_conv(ws + o_t2, 0, nullptr, ws + p_d2b, d2b_b, ws + o_db, 32, 16, NN, 1, 1, stream);
    launch_conv(ws + o_db, 1, ws + o_x0, ws + p_d1a, d1a_b, ws + o_t1, 16, 32, NN, 1, 2, stream);
    launch_conv(ws + o_t1, 0, nullptr, ws + p_d1b, d1b_b, ws + o_g2, 16, 32, NN, 1, 1, stream);
    {
        long tot = 31L * 32768;
        conv1x1_kernel<<<(tot + 255) / 256, 256, 0, stream>>>(ws + o_g2, d1c_w, d1c_b, out);
    }
}
